// PAM_Module_120259084307
// MI455X (gfx1250) — matrix operand order repaired and checked by reading, not compiled, not run on hardware
//
#include <hip/hip_runtime.h>

typedef __attribute__((ext_vector_type(16))) _Float16 v16h;
typedef __attribute__((ext_vector_type(8)))  _Float16 v8h;
typedef __attribute__((ext_vector_type(8)))  float    v8f;

#define BATCH 2
#define CH    64
#define NPIX  6400      // 80*80
#define CQD   8         // query/key channels
#define WPT   8         // waves cooperating on one m-tile (n-stripe split)
#define NSTRIPE (NPIX / WPT)   // 800 columns per wave, 25 chunks of 32

#define LOG2E 1.44269504088896340736f

// ---------------------------------------------------------------------------
// Kernel 1: per-pixel 1x1-conv projections.
//   qt[b][n][0..7]  (f16)  -- stores q * log2(e): QK logits come out of the
//                             WMMA base-2, softmax is a raw v_exp_f32.
//   kt[b][n][0..7]  (f16)
//   vt[b][c][n]     (f16)  -- n-contiguous: PV B-operand is two 16B loads/lane
// ---------------------------------------------------------------------------
__global__ __launch_bounds__(256) void pam_proj_kernel(
    const float* __restrict__ x1, const float* __restrict__ x2,
    const float* __restrict__ Wq, const float* __restrict__ bq,
    const float* __restrict__ Wk, const float* __restrict__ bk,
    const float* __restrict__ Wv, const float* __restrict__ bv,
    _Float16* __restrict__ qt, _Float16* __restrict__ kt,
    _Float16* __restrict__ vt)
{
    int idx = blockIdx.x * 256 + threadIdx.x;
    if (idx >= BATCH * NPIX) return;
    int b = idx / NPIX;
    int n = idx - b * NPIX;

    float xr[CH];
    const float* xp = x1 + (size_t)b * CH * NPIX + n;
#pragma unroll
    for (int c = 0; c < CH; ++c) xr[c] = xp[(size_t)c * NPIX];

    v8h qs;
#pragma unroll
    for (int o = 0; o < CQD; ++o) {
        float s = bq[o];
#pragma unroll
        for (int c = 0; c < CH; ++c) s = fmaf(Wq[o * CH + c], xr[c], s);
        qs[o] = (_Float16)(s * LOG2E);     // base-2 pre-scale
    }
    *(v8h*)(qt + (size_t)idx * CQD) = qs;

#pragma unroll 2
    for (int o = 0; o < CH; ++o) {
        float s = bv[o];
#pragma unroll
        for (int c = 0; c < CH; ++c) s = fmaf(Wv[o * CH + c], xr[c], s);
        vt[((size_t)b * CH + o) * NPIX + n] = (_Float16)s;
    }

    const float* xp2 = x2 + (size_t)b * CH * NPIX + n;
#pragma unroll
    for (int c = 0; c < CH; ++c) xr[c] = xp2[(size_t)c * NPIX];
    v8h ks;
#pragma unroll
    for (int o = 0; o < CQD; ++o) {
        float s = bk[o];
#pragma unroll
        for (int c = 0; c < CH; ++c) s = fmaf(Wk[o * CH + c], xr[c], s);
        ks[o] = (_Float16)s;
    }
    *(v8h*)(kt + (size_t)idx * CQD) = ks;
}

// ---------------------------------------------------------------------------
// Kernel 2: attention. 800 blocks x 8 waves; 8 waves share one 16-row m-tile,
// splitting the n-sweep into 800-column stripes (25 chunks each). exp2 needs
// no max subtraction (|logit| bounded far below f32 overflow; softmax is
// shift-invariant), so partials combine by plain addition in the LDS tail.
// Per chunk (no cross-wave barrier -- only compiler scheduling barriers):
//   S' = (Q*log2e)(16x32, K padded 8->32) x K(32x16) : 2x WMMA f16
//   P  = exp2(S') : 16 independent v_exp_f32
//   P written to LDS DIRECTLY in A-operand layout: 16 independent
//     ds_store_b16 (immediate offsets, no waits), read back as ONE aligned
//     v16h per lane (ds_load_b128 x2, single wait).
//   O += P(16x32) x Vt(32x16) per 16-channel group : 4x WMMA
// ---------------------------------------------------------------------------
__global__ __launch_bounds__(256) void pam_attn_kernel(
    const _Float16* __restrict__ qt,
    const _Float16* __restrict__ kt,
    const _Float16* __restrict__ vt,
    const float* __restrict__ x1,
    const float* __restrict__ gamma,
    float* __restrict__ out)
{
    __shared__ v16h  pabuf[WPT * 32];          // per-wave P tile, A-layout:
                                               // lane l's operand = pabuf[w*32+l]
    __shared__ float accbuf[WPT][32][40];      // [wave][lane][t*8+r], padded
    __shared__ float lbuf[WPT][32][8];         // per-wave row denominators

    const int wave = threadIdx.x >> 5;
    const int lane = threadIdx.x & 31;
    const int lg   = lane >> 4;                // half-wave group (0/1)
    const int ln   = lane & 15;
    const int b    = blockIdx.x / 400;
    const int m0   = (blockIdx.x - b * 400) * 16;
    const int nbeg = wave * NSTRIPE;

    // Q A-operand (16x32 f16, K padded 8->32): real data only in lanes 0-15,
    // elements 0..7; everything else zero. Same tile for all 8 waves.
    v16h qa = {};
    if (lg == 0) {
        v8h q8 = *(const v8h*)(qt + (size_t)(b * NPIX + m0 + ln) * CQD);
#pragma unroll
        for (int e = 0; e < 8; ++e) qa[e] = q8[e];
    }

    v8f acc[4];
#pragma unroll
    for (int t = 0; t < 4; ++t) acc[t] = (v8f){};
    float lsum[8];
#pragma unroll
    for (int r = 0; r < 8; ++r) lsum[r] = 0.f;

    _Float16* pb = (_Float16*)(pabuf + wave * 32);
    // Scatter-store mapping for this producer lane (columns n=ln and n=16+ln):
    //   value P(m, n) -> reader lane lt = m + 16*gt, element e
    //   n<16:  gt = ln>>3, e = ln&7 ; n>=16: same gt, e = 8 + (ln&7)
    const int gt    = ln >> 3;
    const int e0    = ln & 7;
    const int wbase = (8 * lg + 16 * gt) * 16;   // halves; + r*16 per row

    for (int n0 = nbeg; n0 < nbeg + NSTRIPE; n0 += 32) {
        v16h kb0 = {}, kb1 = {};
        if (lg == 0) {
            v8h k8 = *(const v8h*)(kt + (size_t)(b * NPIX + n0 + ln) * CQD);
            v8h k9 = *(const v8h*)(kt + (size_t)(b * NPIX + n0 + 16 + ln) * CQD);
#pragma unroll
            for (int e = 0; e < 8; ++e) { kb0[e] = k8[e]; kb1[e] = k9[e]; }
        }
        v8f S0 = (v8f){}, S1 = (v8f){};
        S0 = __builtin_amdgcn_wmma_f32_16x16x32_f16(false, qa, false, kb0,
                                                    (short)0, S0, false, false);
        S1 = __builtin_amdgcn_wmma_f32_16x16x32_f16(false, qa, false, kb1,
                                                    (short)0, S1, false, false);

        // P = exp2(S'); scatter directly into A-layout slots (no RAW waits).
#pragma unroll
        for (int r = 0; r < 8; ++r) {
            float p0 = __builtin_amdgcn_exp2f(S0[r]);
            float p1 = __builtin_amdgcn_exp2f(S1[r]);
            lsum[r] += p0 + p1;
            pb[wbase + r * 16 + e0]     = (_Float16)p0;
            pb[wbase + r * 16 + 8 + e0] = (_Float16)p1;
        }
        // per-wave LDS is in-order in HW; this only stops compiler reordering
        __builtin_amdgcn_wave_barrier();

        // Whole A-operand in one aligned 32B read per lane.
        v16h pa = pabuf[wave * 32 + lane];
        __builtin_amdgcn_wave_barrier();

#pragma unroll
        for (int t = 0; t < 4; ++t) {
            const _Float16* vp =
                vt + ((size_t)(b * CH + t * 16 + ln)) * NPIX + n0 + lg * 8;
            v8h vlo = *(const v8h*)vp;
            v8h vhi = *(const v8h*)(vp + 16);
            v16h vb = __builtin_shufflevector(vlo, vhi, 0, 1, 2, 3, 4, 5, 6, 7, 8, 9, 10, 11, 12, 13, 14, 15);
            acc[t] = __builtin_amdgcn_wmma_f32_16x16x32_f16(
                false, pa, false, vb, (short)0, acc[t], false, false);
        }
    }

    // Row-reduce this wave's partial denominators (16-lane groups), then
    // publish partials for the cross-wave combine.
#pragma unroll
    for (int r = 0; r < 8; ++r) {
#pragma unroll
        for (int off = 8; off >= 1; off >>= 1)
            lsum[r] += __shfl_xor(lsum[r], off, 16);
    }
#pragma unroll
    for (int t = 0; t < 4; ++t)
        *(v8f*)&accbuf[wave][lane][t * 8] = acc[t];
    v8f ls;
#pragma unroll
    for (int r = 0; r < 8; ++r) ls[r] = lsum[r];
    *(v8f*)&lbuf[wave][lane][0] = ls;
    __syncthreads();

    // Waves 0..3 each combine one 16-channel group and write the output.
    if (wave < 4) {
        const int t = wave;
        v8f a = (v8f){};
        v8f lt = (v8f){};
#pragma unroll
        for (int w = 0; w < WPT; ++w) {
            a  += *(const v8f*)&accbuf[w][lane][t * 8];
            lt += *(const v8f*)&lbuf[w][lane][0];
        }
        // acc elements r are 8 consecutive pixels m = m0+8*lg+r for channel
        // c = t*16+ln -> vectorized 32B residual load + store.
        float g = gamma[0];
        size_t base = ((size_t)(b * CH + t * 16 + ln)) * NPIX + m0 + lg * 8;
        v8f xres = *(const v8f*)(x1 + base);
        v8f o;
#pragma unroll
        for (int r = 0; r < 8; ++r)
            o[r] = g * (a[r] / lt[r]) + xres[r];
        *(v8f*)(out + base) = o;
    }
}

// ---------------------------------------------------------------------------
extern "C" void kernel_launch(void* const* d_in, const int* in_sizes, int n_in,
                              void* d_out, int out_size, void* d_ws, size_t ws_size,
                              hipStream_t stream) {
    const float* x1    = (const float*)d_in[0];
    const float* x2    = (const float*)d_in[1];
    const float* Wq    = (const float*)d_in[2];
    const float* bq    = (const float*)d_in[3];
    const float* Wk    = (const float*)d_in[4];
    const float* bk    = (const float*)d_in[5];
    const float* Wv    = (const float*)d_in[6];
    const float* bv    = (const float*)d_in[7];
    const float* gamma = (const float*)d_in[8];
    float* out = (float*)d_out;

    // workspace (256B-aligned slabs): qt 204800B | kt 204800B | vt 1638400B
    _Float16* qt = (_Float16*)d_ws;
    _Float16* kt = (_Float16*)((char*)d_ws + 204800);
    _Float16* vt = (_Float16*)((char*)d_ws + 409600);

    pam_proj_kernel<<<(BATCH * NPIX + 255) / 256, 256, 0, stream>>>(
        x1, x2, Wq, bq, Wk, bk, Wv, bv, qt, kt, vt);

    // 800 m-tiles x 8 cooperating waves = 6400 waves in flight
    pam_attn_kernel<<<800, 256, 0, stream>>>(qt, kt, vt, x1, gamma, out);
}
